// Attention_266287972829
// MI455X (gfx1250) — matrix operand order repaired and checked by reading, not compiled, not run on hardware
//
#include <hip/hip_runtime.h>
#include <math.h>

typedef __attribute__((ext_vector_type(16))) _Float16 v16h;
typedef __attribute__((ext_vector_type(8)))  float    v8f;
typedef __attribute__((ext_vector_type(4)))  float    v4f;

#define SEQ  4096
#define HDIM 32
#define DDIM 16
#define EPS      1e-8f
#define RCP_EPS  1e8f   // 1/EPS : clamp for rsq to emulate max(norm, EPS)

// ---------------------------------------------------------------------------
// Kernel 1: kn[b, :] = normalize(dec[b, :] @ Wk^T + bk)   (tiny: 512 x 16)
// ---------------------------------------------------------------------------
__global__ void kn_kernel(const float* __restrict__ dec,
                          const float* __restrict__ Wk,
                          const float* __restrict__ bk,
                          float* __restrict__ kn, int B) {
    int b = blockIdx.x * blockDim.x + threadIdx.x;
    if (b >= B) return;
    float h[HDIM];
#pragma unroll
    for (int i = 0; i < HDIM; ++i) h[i] = dec[b * HDIM + i];
    float kv[DDIM];
    float n2 = 0.f;
#pragma unroll
    for (int d = 0; d < DDIM; ++d) {
        float s = bk[d];
#pragma unroll
        for (int i = 0; i < HDIM; ++i) s += h[i] * Wk[d * HDIM + i];
        kv[d] = s;
        n2 += s * s;
    }
    float denom = fmaxf(sqrtf(n2), EPS);
#pragma unroll
    for (int d = 0; d < DDIM; ++d) kn[b * DDIM + d] = kv[d] / denom;
}

// ---------------------------------------------------------------------------
// Kernel 2: per batch row b (one WG, 8 waves):
//   qT_tile = Wq(16x32) x enc_tile^T(32x16)      -- v_wmma_f32_16x16x32_f16
//   D-layout: lane -> s-row (N), vgpr j -> q component d = 8*half + j (M)
//   att[s]  = (q.kn) * min(rsq(|q|^2), 1e8)      -- in-register reduction
//   out[b,s]= log_softmax_s(att)[s]
// ---------------------------------------------------------------------------
__global__ __launch_bounds__(256) void attn_kernel(
        const float* __restrict__ enc,
        const float* __restrict__ Wq,
        const float* __restrict__ bq,
        const float* __restrict__ kn,
        float* __restrict__ out) {
    __shared__ float satt[SEQ];
    __shared__ float sred[8];

    const int b      = blockIdx.x;
    const int tid    = threadIdx.x;
    const int lane   = tid & 31;
    const int w      = tid >> 5;       // wave id 0..7
    const int lane16 = lane & 15;
    const int half   = lane >> 4;      // 0 or 1

    // --- A fragment (loop-invariant): Wq (16x32, M=D rows, K=H) as f16.
    //     Lane -> row M=lane16; half 0: K = 0-15; half 1: K = 16-31.
    v16h afr;
    {
        const float* wr = Wq + lane16 * HDIM;
        const int kb = half * 16;
#pragma unroll
        for (int j = 0; j < 8; ++j) {
            afr[j]     = (_Float16)wr[kb + j];
            afr[8 + j] = (_Float16)wr[kb + 8 + j];
        }
    }
    // Per-lane slice of the D dimension: this lane accumulates d = 8*half + j.
    float bqh[8], knh[8];
#pragma unroll
    for (int j = 0; j < 8; ++j) {
        bqh[j] = bq[8 * half + j];
        knh[j] = kn[b * DDIM + 8 * half + j];
    }

    const float* encb = enc + (size_t)b * SEQ * HDIM;

    // 256 tiles of 16 rows; wave w takes tiles w, w+8, w+16, ...
    for (int t = w; t < SEQ / 16; t += 8) {
        // --- B fragment: enc_tile^T (32x16) as f16.
        //     Lane -> column N = s-row (t*16+lane16);
        //     lanes 0-15 hold K=0..15, lanes 16-31 hold K=16..31
        //     -> one contiguous 64B read per lane (full rows consumed).
        const float* rp = encb + (t * 16 + lane16) * HDIM + half * 16;
        v4f f0 = __builtin_nontemporal_load((const v4f*)(rp));
        v4f f1 = __builtin_nontemporal_load((const v4f*)(rp + 4));
        v4f f2 = __builtin_nontemporal_load((const v4f*)(rp + 8));
        v4f f3 = __builtin_nontemporal_load((const v4f*)(rp + 12));
        // Speculative prefetch of this wave's next tile; address wrapped inside
        // the same batch row so it is always valid (no EXEC-mask branch).
        __builtin_prefetch(
            (const void*)(encb + ((((t + 8) * 16) & (SEQ - 1)) + lane16) * HDIM),
            0, 1);

        v16h bfr;
#pragma unroll
        for (int j = 0; j < 4; ++j) {
            bfr[j]      = (_Float16)f0[j];
            bfr[4 + j]  = (_Float16)f1[j];
            bfr[8 + j]  = (_Float16)f2[j];
            bfr[12 + j] = (_Float16)f3[j];
        }

        v8f c = {};
        v8f d = __builtin_amdgcn_wmma_f32_16x16x32_f16(
                    false, afr, false, bfr, (short)0, c, false, false);
        // d[j] = q[s = t*16+lane16][d = 8*half + j]   (pre-bias)

        float s1 = 0.f, s2 = 0.f;                 // |q|^2 , q.kn (half each)
#pragma unroll
        for (int j = 0; j < 8; ++j) {
            float q = d[j] + bqh[j];
            s1 = fmaf(q, q, s1);
            s2 = fmaf(q, knh[j], s2);
        }
        // lane ^ 16 holds the other 8 components of the same s-row
        s1 += __shfl_xor(s1, 16, 32);
        s2 += __shfl_xor(s2, 16, 32);

        float att = s2 * fminf(__builtin_amdgcn_rsqf(s1), RCP_EPS);
        if (half == 0) satt[t * 16 + lane16] = att;   // 16 contiguous floats
    }
    __syncthreads();

    // ---- log-softmax over S ----
    float lm = -INFINITY;
    for (int i = tid; i < SEQ; i += 256) lm = fmaxf(lm, satt[i]);
#pragma unroll
    for (int m = 16; m >= 1; m >>= 1) lm = fmaxf(lm, __shfl_xor(lm, m, 32));
    if (lane == 0) sred[w] = lm;
    __syncthreads();
    float bm = sred[0];
#pragma unroll
    for (int i = 1; i < 8; ++i) bm = fmaxf(bm, sred[i]);

    float ls = 0.f;
    for (int i = tid; i < SEQ; i += 256) ls += expf(satt[i] - bm);
#pragma unroll
    for (int m = 16; m >= 1; m >>= 1) ls += __shfl_xor(ls, m, 32);
    __syncthreads();               // done reading sred from the max phase
    if (lane == 0) sred[w] = ls;
    __syncthreads();
    float bs = 0.f;
#pragma unroll
    for (int i = 0; i < 8; ++i) bs += sred[i];

    const float lse = bm + logf(bs);
    float* outb = out + (size_t)b * SEQ;
    for (int i = tid; i < SEQ; i += 256)
        __builtin_nontemporal_store(satt[i] - lse, outb + i);
}

// ---------------------------------------------------------------------------
extern "C" void kernel_launch(void* const* d_in, const int* in_sizes, int n_in,
                              void* d_out, int out_size, void* d_ws, size_t ws_size,
                              hipStream_t stream) {
    const float* dec = (const float*)d_in[0];  // [B, 32]
    const float* enc = (const float*)d_in[1];  // [B, 4096, 32]
    const float* Wq  = (const float*)d_in[2];  // [16, 32]
    const float* bq  = (const float*)d_in[3];  // [16]
    const float* Wk  = (const float*)d_in[4];  // [16, 32]
    const float* bk  = (const float*)d_in[5];  // [16]
    float*       out = (float*)d_out;          // [B, 1, 4096]
    float*       kn  = (float*)d_ws;           // [B, 16] scratch

    const int B = in_sizes[0] / HDIM;          // 512

    kn_kernel<<<(B + 255) / 256, 256, 0, stream>>>(dec, Wk, bk, kn, B);
    attn_kernel<<<B, 256, 0, stream>>>(enc, Wq, bq, kn, out);
}
